// ChannelAttentionBlock_70909910057219
// MI455X (gfx1250) — matrix operand order repaired and checked by reading, not compiled, not run on hardware
//
#include <hip/hip_runtime.h>

// ---------------------------------------------------------------------------
// ChannelAttentionBlock for MI455X (gfx1250, wave32, WMMA bf16 + TDM staging)
//   B=16, C=512, H=W=64, HW=4096
// ---------------------------------------------------------------------------

#define BATCH 16
#define CDIM  512
#define HWDIM 4096

typedef __attribute__((ext_vector_type(16))) __bf16       v16bf;
typedef __attribute__((ext_vector_type(8)))  __bf16       v8bf;
typedef __attribute__((ext_vector_type(8)))  float        v8f;
typedef __attribute__((ext_vector_type(4)))  unsigned int v4u;
typedef __attribute__((ext_vector_type(8)))  int          v8i;
typedef __attribute__((ext_vector_type(4)))  int          v4i;

// LDS panel geometry: K-step = 32 bf16 = 64B data per row, padded to 80B by the
// TDM (pad 16B every 64B) so that 16 consecutive rows hit distinct LDS banks.
#define KSTEP          32
#define LDS_ROW_BYTES  80
#define A_ROWS         128
#define B_ROWS         256
#define A_PANEL_BYTES  (A_ROWS * LDS_ROW_BYTES)          // 10240
#define B_PANEL_BYTES  (B_ROWS * LDS_ROW_BYTES)          // 20480
#define LDS_TOTAL      (2 * A_PANEL_BYTES + 2 * B_PANEL_BYTES)  // 61440

// The TDM writes LDS behind the compiler's back (the builtin carries only
// descriptor values, no LDS pointer). Two pieces are needed so the compiler
// cannot fold the fragment loads to undef:
//  1) escape the LDS pointer into opaque asm once (defeats capture analysis /
//     GlobalOpt's "internal global with no stores" rewrite), and
//  2) a per-iteration memory clobber so loads re-read after each TDM arrival.
__device__ __forceinline__ void lds_escape(void* p) {
    asm volatile("" : : "v"(p));
}
__device__ __forceinline__ void lds_publish() { asm volatile("" ::: "memory"); }

// ---------------------------------------------------------------------------
// Tensor Data Mover: stage a [tile_rows x 32] bf16 tile (row stride
// row_elems) from global memory into LDS at byte offset lds_off, with LDS
// padding 64B->80B rows. D# layout per CDNA5 ISA.
// Must be executed by one full wave; tracked with TENSORcnt.
// This toolchain uses the 6-arg builtin:
//   (uint32x4 g0, int32x8 g1, int32x4 g2, int32x4 g3, int32x8 extra, i32 cpol)
// ---------------------------------------------------------------------------
__device__ __forceinline__ void
tdm_stage_2d(unsigned lds_off, const __bf16* gsrc,
             unsigned tile_rows, unsigned row_elems) {
    const unsigned long long ga = (unsigned long long)gsrc;

    v4u g0;
    g0[0] = 1u;                                     // count=1, user mode
    g0[1] = lds_off;                                // lds_addr (bytes)
    g0[2] = (unsigned)ga;                           // global_addr[31:0]
    g0[3] = (unsigned)((ga >> 32) & 0x01FFFFFFu)    // global_addr[56:32]
          | (2u << 30);                             // type = 2 ("image")

    const unsigned dim0 = row_elems;                // tensor width (elements)
    const unsigned dim1 = tile_rows;                // rows available
    v8i g1;
    g1[0] = (int)((1u << 16)                        // data_size = 2 bytes
                | (1u << 20)                        // pad_enable
                | (3u << 22)                        // pad_interval: 16 DW (64B)
                | (3u << 25));                      // pad_amount:   4 DW (16B)
    g1[1] = (int)((dim0 & 0xFFFFu) << 16);          // tensor_dim0[15:0]
    g1[2] = (int)(((dim0 >> 16) & 0xFFFFu)          // tensor_dim0[31:16]
                | ((dim1 & 0xFFFFu) << 16));        // tensor_dim1[15:0]
    g1[3] = (int)(((dim1 >> 16) & 0xFFFFu)          // tensor_dim1[31:16]
                | ((unsigned)KSTEP << 16));         // tile_dim0 = 32
    g1[4] = (int)(tile_rows & 0xFFFFu);             // tile_dim1; tile_dim2 = 0
    g1[5] = (int)row_elems;                         // tensor_dim0_stride[31:0]
    g1[6] = 0;                                      // stride hi / dim1_stride lo
    g1[7] = 0;

    const v4i z4 = {0, 0, 0, 0};                    // groups 2/3: 2-D tile
    const v8i z8 = {0, 0, 0, 0, 0, 0, 0, 0};
    __builtin_amdgcn_tensor_load_to_lds(g0, g1, z4, z4, z8, 0);
}

// ---------------------------------------------------------------------------
// WMMA fragment gather from an LDS panel (padded 80B rows).
//  A (16x32 bf16): lane m=lane&15, kh=lane>>4; K chunks kh*8 and 16+kh*8
//  B (32x16 bf16): lane n=lane&15, kh=lane>>4; K chunks kh*8 and 16+kh*8
// ---------------------------------------------------------------------------
__device__ __forceinline__ v16bf combine16(v8bf lo, v8bf hi) {
    return __builtin_shufflevector(lo, hi, 0, 1, 2, 3, 4, 5, 6, 7,
                                           8, 9, 10, 11, 12, 13, 14, 15);
}
__device__ __forceinline__ v16bf lds_a_frag(const char* panel, int row, int kh) {
    const char* r = panel + row * LDS_ROW_BYTES;
    v8bf lo = *(const v8bf*)(r + kh * 16);
    v8bf hi = *(const v8bf*)(r + 32 + kh * 16);
    return combine16(lo, hi);
}
__device__ __forceinline__ v16bf lds_b_frag(const char* panel, int row, int kh) {
    const char* r = panel + row * LDS_ROW_BYTES;
    v8bf lo = *(const v8bf*)(r + kh * 16);
    v8bf hi = *(const v8bf*)(r + 32 + kh * 16);
    return combine16(lo, hi);
}

// ---------------------------------------------------------------------------
// Kernel 1: f32 -> bf16 convert, x_bf16 [B,C,HW] and xT_bf16 [B,HW,C]
// ---------------------------------------------------------------------------
__global__ void __launch_bounds__(256)
cvt_transpose_kernel(const float* __restrict__ x,
                     __bf16* __restrict__ xb,
                     __bf16* __restrict__ xt) {
    __shared__ __bf16 tile[64][65];
    const int b  = blockIdx.z;
    const int c0 = blockIdx.y * 64;
    const int n0 = blockIdx.x * 64;
    const float* xp  = x  + (size_t)b * CDIM * HWDIM;
    __bf16*      xbp = xb + (size_t)b * CDIM * HWDIM;
    __bf16*      xtp = xt + (size_t)b * HWDIM * CDIM;
    const int t = threadIdx.x;

#pragma unroll
    for (int i = 0; i < 16; ++i) {
        const int idx = t + i * 256;
        const int r   = idx >> 6;
        const int cc  = idx & 63;
        const float v = xp[(size_t)(c0 + r) * HWDIM + (n0 + cc)];
        const __bf16 bv = (__bf16)v;
        xbp[(size_t)(c0 + r) * HWDIM + (n0 + cc)] = bv;
        tile[cc][r] = bv;
    }
    __syncthreads();
#pragma unroll
    for (int i = 0; i < 16; ++i) {
        const int idx = t + i * 256;
        const int r   = idx >> 6;
        const int cc  = idx & 63;
        xtp[(size_t)(n0 + r) * CDIM + (c0 + cc)] = tile[r][cc];
    }
}

// ---------------------------------------------------------------------------
// Kernel 2: scores = x . x^T   (per batch M=N=512, K=4096)
// Workgroup tile 128(M) x 256(N); 8 waves as 2x4; wave tile 64x64 (4x4 WMMA).
// TDM double-buffered LDS staging of A[128x32] and B[256x32] panels.
// ---------------------------------------------------------------------------
__global__ void __launch_bounds__(256)
gemm_scores_kernel(const __bf16* __restrict__ xb, float* __restrict__ sc) {
    __shared__ char smem[LDS_TOTAL];
    lds_escape(smem);                             // TDM writes smem: make it escape

    const int b    = blockIdx.z;
    const int m0   = blockIdx.y * 128;
    const int n0   = blockIdx.x * 256;
    const int w    = threadIdx.x >> 5;
    const int lane = threadIdx.x & 31;
    const int wm   = (w & 1) * 64;
    const int wn   = (w >> 1) * 64;
    const int lm   = lane & 15;
    const int kh   = lane >> 4;

    const __bf16* xbase = xb + (size_t)b * CDIM * HWDIM;
    const __bf16* Asrc  = xbase + (size_t)m0 * HWDIM;
    const __bf16* Bsrc  = xbase + (size_t)n0 * HWDIM;

    const unsigned aoff[2] = {0u, A_PANEL_BYTES};
    const unsigned boff[2] = {2u * A_PANEL_BYTES, 2u * A_PANEL_BYTES + B_PANEL_BYTES};

    if (threadIdx.x < 32) {                       // wave 0 drives the TDM
        tdm_stage_2d(aoff[0], Asrc, A_ROWS, HWDIM);
        tdm_stage_2d(boff[0], Bsrc, B_ROWS, HWDIM);
    }

    v8f acc[4][4];
#pragma unroll
    for (int i = 0; i < 4; ++i)
#pragma unroll
        for (int j = 0; j < 4; ++j) acc[i][j] = (v8f){};

    int buf = 0;
    for (int kk = 0; kk < HWDIM; kk += KSTEP) {
        if (threadIdx.x < 32) __builtin_amdgcn_s_wait_tensorcnt(0);
        __syncthreads();                          // panel ready; prev reads done
        lds_publish();                            // TDM wrote LDS: force reloads
        const int nb = buf ^ 1;
        if (kk + KSTEP < HWDIM && threadIdx.x < 32) {
            tdm_stage_2d(aoff[nb], Asrc + kk + KSTEP, A_ROWS, HWDIM);
            tdm_stage_2d(boff[nb], Bsrc + kk + KSTEP, B_ROWS, HWDIM);
        }
        if (threadIdx.x == 0 && kk + 2 * KSTEP < HWDIM) {
            __builtin_prefetch(Asrc + kk + 2 * KSTEP, 0, 3);
            __builtin_prefetch(Bsrc + kk + 2 * KSTEP, 0, 3);
        }

        const char* Ap = smem + aoff[buf];
        const char* Bp = smem + boff[buf];
        v16bf a[4], bm[4];
#pragma unroll
        for (int i = 0; i < 4; ++i) a[i]  = lds_a_frag(Ap, wm + i * 16 + lm, kh);
#pragma unroll
        for (int j = 0; j < 4; ++j) bm[j] = lds_b_frag(Bp, wn + j * 16 + lm, kh);
#pragma unroll
        for (int i = 0; i < 4; ++i)
#pragma unroll
            for (int j = 0; j < 4; ++j)
                acc[i][j] = __builtin_amdgcn_wmma_f32_16x16x32_bf16(
                    false, a[i], false, bm[j], (short)0, acc[i][j], false, false);
        buf = nb;
    }

    float* srow = sc + (size_t)b * CDIM * CDIM;
#pragma unroll
    for (int i = 0; i < 4; ++i)
#pragma unroll
        for (int j = 0; j < 4; ++j)
#pragma unroll
            for (int v = 0; v < 8; ++v) {
                const int m = m0 + wm + i * 16 + v + kh * 8;
                const int n = n0 + wn + j * 16 + lm;
                srow[(size_t)m * CDIM + n] = acc[i][j][v];
            }
}

// ---------------------------------------------------------------------------
// Kernel 3: softmax(rowmax - scores):  attn_d = exp(rowmin - s_d)/sum
// One wave32 per row; bf16 output.
// ---------------------------------------------------------------------------
__global__ void __launch_bounds__(256)
softmax_rows_kernel(const float* __restrict__ sc, __bf16* __restrict__ attn) {
    const int row  = blockIdx.x * 8 + (threadIdx.x >> 5);
    const int lane = threadIdx.x & 31;
    const float* sp = sc + (size_t)row * CDIM;

    float vals[16];
    float vmin = 3.402823466e+38f;
#pragma unroll
    for (int i = 0; i < 16; ++i) {
        vals[i] = sp[lane + i * 32];
        vmin = fminf(vmin, vals[i]);
    }
#pragma unroll
    for (int off = 16; off > 0; off >>= 1)
        vmin = fminf(vmin, __shfl_xor(vmin, off, 32));

    float sum = 0.0f;
#pragma unroll
    for (int i = 0; i < 16; ++i) {
        vals[i] = __expf(vmin - vals[i]);
        sum += vals[i];
    }
#pragma unroll
    for (int off = 16; off > 0; off >>= 1)
        sum += __shfl_xor(sum, off, 32);

    const float inv = 1.0f / sum;
    __bf16* ap = attn + (size_t)row * CDIM;
#pragma unroll
    for (int i = 0; i < 16; ++i)
        ap[lane + i * 32] = (__bf16)(vals[i] * inv);
}

// ---------------------------------------------------------------------------
// Kernel 4: out = gamma * (attn . x) + inputs
//   per batch M=512 (c), N=4096 (n), K=512 (d); A=attn bf16, B=xT bf16.
// Same TDM-staged structure as kernel 2; fused f32 epilogue.
// ---------------------------------------------------------------------------
__global__ void __launch_bounds__(256)
gemm_out_kernel(const __bf16* __restrict__ attn,
                const __bf16* __restrict__ xt,
                const float* __restrict__ xin,
                const float* __restrict__ gamma,
                float* __restrict__ out) {
    __shared__ char smem[LDS_TOTAL];
    lds_escape(smem);                             // TDM writes smem: make it escape

    const int b    = blockIdx.z;
    const int m0   = blockIdx.y * 128;            // channel tile
    const int n0   = blockIdx.x * 256;            // spatial tile
    const int w    = threadIdx.x >> 5;
    const int lane = threadIdx.x & 31;
    const int wm   = (w & 1) * 64;
    const int wn   = (w >> 1) * 64;
    const int lm   = lane & 15;
    const int kh   = lane >> 4;

    const __bf16* Asrc = attn + (size_t)b * CDIM * CDIM + (size_t)m0 * CDIM;
    const __bf16* Bsrc = xt + (size_t)b * HWDIM * CDIM + (size_t)n0 * CDIM;

    const unsigned aoff[2] = {0u, A_PANEL_BYTES};
    const unsigned boff[2] = {2u * A_PANEL_BYTES, 2u * A_PANEL_BYTES + B_PANEL_BYTES};

    if (threadIdx.x < 32) {
        tdm_stage_2d(aoff[0], Asrc, A_ROWS, CDIM);
        tdm_stage_2d(boff[0], Bsrc, B_ROWS, CDIM);
    }

    v8f acc[4][4];
#pragma unroll
    for (int i = 0; i < 4; ++i)
#pragma unroll
        for (int j = 0; j < 4; ++j) acc[i][j] = (v8f){};

    int buf = 0;
    for (int kk = 0; kk < CDIM; kk += KSTEP) {
        if (threadIdx.x < 32) __builtin_amdgcn_s_wait_tensorcnt(0);
        __syncthreads();
        lds_publish();                            // TDM wrote LDS: force reloads
        const int nb = buf ^ 1;
        if (kk + KSTEP < CDIM && threadIdx.x < 32) {
            tdm_stage_2d(aoff[nb], Asrc + kk + KSTEP, A_ROWS, CDIM);
            tdm_stage_2d(boff[nb], Bsrc + kk + KSTEP, B_ROWS, CDIM);
        }

        const char* Ap = smem + aoff[buf];
        const char* Bp = smem + boff[buf];
        v16bf a[4], bm[4];
#pragma unroll
        for (int i = 0; i < 4; ++i) a[i]  = lds_a_frag(Ap, wm + i * 16 + lm, kh);
#pragma unroll
        for (int j = 0; j < 4; ++j) bm[j] = lds_b_frag(Bp, wn + j * 16 + lm, kh);
#pragma unroll
        for (int i = 0; i < 4; ++i)
#pragma unroll
            for (int j = 0; j < 4; ++j)
                acc[i][j] = __builtin_amdgcn_wmma_f32_16x16x32_bf16(
                    false, a[i], false, bm[j], (short)0, acc[i][j], false, false);
        buf = nb;
    }

    const float g = gamma[0];
    const float* xrow = xin + (size_t)b * CDIM * HWDIM;
    float*       orow = out + (size_t)b * CDIM * HWDIM;
#pragma unroll
    for (int i = 0; i < 4; ++i)
#pragma unroll
        for (int j = 0; j < 4; ++j)
#pragma unroll
            for (int v = 0; v < 8; ++v) {
                const int m = m0 + wm + i * 16 + v + kh * 8;
                const int n = n0 + wn + j * 16 + lm;
                const size_t idx = (size_t)m * HWDIM + n;
                orow[idx] = g * acc[i][j][v] + xrow[idx];
            }
}

// ---------------------------------------------------------------------------
// Launch: ws layout
//   [0, 64MiB)        x_bf16  [B,C,HW]   (overlaid later by attn bf16, 8MiB)
//   [64MiB, 128MiB)   xT_bf16 [B,HW,C]
//   [128MiB, 144MiB)  scores f32 [B,C,C]
// ---------------------------------------------------------------------------
extern "C" void kernel_launch(void* const* d_in, const int* in_sizes, int n_in,
                              void* d_out, int out_size, void* d_ws, size_t ws_size,
                              hipStream_t stream) {
    const float* x     = (const float*)d_in[0];
    const float* gamma = (const float*)d_in[1];
    float*       out   = (float*)d_out;

    char* ws = (char*)d_ws;
    __bf16* xb = (__bf16*)ws;
    __bf16* xt = (__bf16*)(ws + (size_t)67108864);
    float*  sc = (float*)(ws + (size_t)134217728);
    __bf16* at = (__bf16*)ws;                     // reuse xb region after GEMM1

    cvt_transpose_kernel<<<dim3(HWDIM / 64, CDIM / 64, BATCH), 256, 0, stream>>>(x, xb, xt);
    gemm_scores_kernel<<<dim3(CDIM / 256, CDIM / 128, BATCH), 256, 0, stream>>>(xb, sc);
    softmax_rows_kernel<<<(BATCH * CDIM) / 8, 256, 0, stream>>>(sc, at);
    gemm_out_kernel<<<dim3(HWDIM / 256, CDIM / 128, BATCH), 256, 0, stream>>>(at, xt, x, gamma, out);
}
